// ComponentNN_53403623358539
// MI455X (gfx1250) — matrix operand order repaired and checked by reading, not compiled, not run on hardware
//
#include <hip/hip_runtime.h>
#include <stdint.h>
#include <stddef.h>

// ---------------------------------------------------------------------------
// MoE soft-blend MLP for MI455X (gfx1250): wave32, bf16 WMMA 16x16x32 (f32 acc),
// double-buffered async global->LDS staging so the matrix pipe overlaps copies.
// Block = 8 waves = 128M x 64N tile; all waves share the staged B (weight) tile.
// ---------------------------------------------------------------------------

typedef __attribute__((ext_vector_type(16))) __bf16 v16bf;
typedef __attribute__((ext_vector_type(8)))  float  v8f;
typedef int v4i_t __attribute__((vector_size(16)));

#define B_ROWS 4096
#define NEXP   8
#define KC     64      // K elements staged per chunk
#define LDST   72      // LDS row stride in elements (144B = 16B-aligned, bank-spread)

#if defined(__AMDGCN__) && __has_builtin(__builtin_amdgcn_global_load_async_to_lds_b128) && __has_builtin(__builtin_amdgcn_s_wait_asynccnt)
#define USE_ASYNC 1
#pragma message("gfx1250: using global_load_async_to_lds_b128 + s_wait_asynccnt")
#else
#define USE_ASYNC 0
#pragma message("gfx1250: async-LDS builtins unavailable; sync load + ds_store staging")
#endif

#if USE_ASYNC
#define WAIT_ASYNC(n) __builtin_amdgcn_s_wait_asynccnt(n)
typedef __attribute__((address_space(1))) v4i_t* gptr4;
typedef __attribute__((address_space(3))) v4i_t* lptr4;
#else
#define WAIT_ASYNC(n) do {} while (0)
#endif

struct Frag { union { v16bf v; uint4 u[2]; }; };

static __device__ __forceinline__ void async_copy16(const unsigned short* gsrc,
                                                    unsigned short* ldst) {
#if USE_ASYNC
  __builtin_amdgcn_global_load_async_to_lds_b128(
      (gptr4)gsrc, (lptr4)ldst, 0, 0);
#else
  uint4 v = *(const uint4*)gsrc;
  *(uint4*)ldst = v;
#endif
}

static __device__ __forceinline__ unsigned short f32_to_bf16_rne(float f) {
  unsigned int u = __float_as_uint(f);
  u += 0x7FFFu + ((u >> 16) & 1u);
  return (unsigned short)(u >> 16);
}

__global__ __launch_bounds__(256) void cvt_f32_to_bf16(
    const float* __restrict__ in, unsigned short* __restrict__ out, int nq) {
  int i = blockIdx.x * blockDim.x + threadIdx.x;
  if (i >= nq) return;
  int idx = i * 4;
  float4 f = *(const float4*)(in + idx);
  uint2 o;
  o.x = (unsigned)f32_to_bf16_rne(f.x) | ((unsigned)f32_to_bf16_rne(f.y) << 16);
  o.y = (unsigned)f32_to_bf16_rne(f.z) | ((unsigned)f32_to_bf16_rne(f.w) << 16);
  *(uint2*)(out + idx) = o;
}

static __device__ __forceinline__ void wmma4(v8f (&acc)[4], const Frag& a,
                                             const Frag (&b)[4]) {
  #pragma unroll
  for (int t = 0; t < 4; ++t)
    acc[t] = __builtin_amdgcn_wmma_f32_16x16x32_bf16(
        false, a.v, false, b[t].v, (short)0, acc[t], false, false);
}

template <int K, int OUT, bool ACT, bool F32OUT>
__global__ __launch_bounds__(256) void moe_layer(
    const unsigned short* __restrict__ Hbf,   // [B_ROWS, K]   bf16
    const unsigned short* __restrict__ Wbf,   // [E, OUT, K]   bf16 (a[e,o,i])
    const float* __restrict__ wb,             // [E, B_ROWS]
    const float* __restrict__ bias,           // [E, OUT]
    unsigned short* __restrict__ outBf,       // [B_ROWS, OUT] bf16 (if !F32OUT)
    float* __restrict__ outF32)               // [B_ROWS, OUT] f32  (if F32OUT)
{
  __shared__ __align__(16) unsigned short lsA[2][128 * LDST];  // 36 KB
  __shared__ __align__(16) unsigned short lsB[2][64 * LDST];   // 18 KB

  const int tid  = threadIdx.x;
  const int wave = tid >> 5;
  const int lane = tid & 31;
  const int bm = blockIdx.x & 31;           // 32 M-blocks of 128 rows
  const int bn = blockIdx.x >> 5;           // N-tile of 64 cols (shared by block)
  const int rowBase = bm * 128;
  const int row0 = rowBase + wave * 16;
  const int col0 = bn * 64;
  const int hi = lane >> 4;
  const int ml = lane & 15;

  // staging decomposition: 256 threads x b128; rr = row-in-group, seg = 8-elem seg
  const int rr  = tid >> 3;                 // 0..31
  const int seg = (tid & 7) * 8;            // 0..56

  // stage one KC-chunk of A (128 rows) and B (64 rows) into buffer p
  auto stage = [&](int p, int e, int kc) {
    #pragma unroll
    for (int j = 0; j < 2; ++j) {           // B: 64 weight rows
      int o = rr + j * 32;
      async_copy16(Wbf + ((size_t)e * OUT + col0 + o) * K + kc + seg,
                   &lsB[p][o * LDST + seg]);
    }
    #pragma unroll
    for (int j = 0; j < 4; ++j) {           // A: 128 activation rows
      int m = rr + j * 32;
      async_copy16(Hbf + (size_t)(rowBase + m) * K + kc + seg,
                   &lsA[p][m * LDST + seg]);
    }
  };

  v8f total[4];
  #pragma unroll
  for (int t = 0; t < 4; ++t) total[t] = v8f{};

  for (int e = 0; e < NEXP; ++e) {
    v8f acc[4];
    #pragma unroll
    for (int t = 0; t < 4; ++t) acc[t] = v8f{};

    int p = 0;
    for (int kc = 0; kc < K; kc += KC) {
      const bool hasNext = (kc + KC) < K;
      __syncthreads();                      // all readers of target buffers done
      if (kc == 0) stage(p, e, 0);          // expert-start: fill current buffer
      if (hasNext) {
        stage(p ^ 1, e, kc + KC);           // prefetch next chunk
        WAIT_ASYNC(6);                      // current chunk landed; next in flight
      } else {
        WAIT_ASYNC(0);
      }
      __syncthreads();                      // whole block sees staged data

      const unsigned short* As = lsA[p];
      const unsigned short* Bs = lsB[p];
      #pragma unroll
      for (int kk = 0; kk < KC; kk += 32) {
        Frag a;
        const unsigned short* ab = As + (wave * 16 + ml) * LDST + kk + hi * 8;
        a.u[0] = *(const uint4*)ab;
        a.u[1] = *(const uint4*)(ab + 16);
        Frag b[4];
        #pragma unroll
        for (int t = 0; t < 4; ++t) {
          const unsigned short* bb = Bs + (t * 16 + ml) * LDST + kk + hi * 8;
          b[t].u[0] = *(const uint4*)bb;
          b[t].u[1] = *(const uint4*)(bb + 16);
        }
        wmma4(acc, a, b);
      }
      p ^= 1;
    }

    // C/D layout: VGPR r of lane (hi,ml) holds M = r + 8*hi, N = ml.
    float wbv[8];
    #pragma unroll
    for (int r = 0; r < 8; ++r)
      wbv[r] = wb[(size_t)e * B_ROWS + row0 + hi * 8 + r];

    #pragma unroll
    for (int t = 0; t < 4; ++t) {
      float bv = bias[(size_t)e * OUT + col0 + t * 16 + ml];
      #pragma unroll
      for (int r = 0; r < 8; ++r)
        total[t][r] += wbv[r] * (acc[t][r] + bv);
    }
  }

  #pragma unroll
  for (int t = 0; t < 4; ++t) {
    #pragma unroll
    for (int r = 0; r < 8; ++r) {
      float v = total[t][r];
      if (ACT) v = (v > 0.f) ? v : (__expf(v) - 1.f);   // ELU (alpha=1)
      const size_t orow = (size_t)(row0 + r + hi * 8);
      const size_t ocol = (size_t)(col0 + t * 16 + ml);
      if (F32OUT) outF32[orow * OUT + ocol] = v;
      else        outBf[orow * OUT + ocol] = f32_to_bf16_rne(v);
    }
  }
}

extern "C" void kernel_launch(void* const* d_in, const int* in_sizes, int n_in,
                              void* d_out, int out_size, void* d_ws, size_t ws_size,
                              hipStream_t stream) {
  (void)in_sizes; (void)n_in; (void)out_size; (void)ws_size;
  const float* x    = (const float*)d_in[0];   // [4096, 512]
  const float* wb   = (const float*)d_in[1];   // [8, 4096]
  const float* a0   = (const float*)d_in[2];   // [8, 1024, 512]
  const float* b0   = (const float*)d_in[3];   // [8, 1024]
  const float* a1   = (const float*)d_in[4];   // [8, 1024, 1024]
  const float* b1   = (const float*)d_in[5];   // [8, 1024]
  const float* a2   = (const float*)d_in[6];   // [8, 512, 1024]
  const float* b2   = (const float*)d_in[7];   // [8, 512]
  float* out = (float*)d_out;                  // [4096, 512]

  char* ws = (char*)d_ws;                      // bf16 buffers, ~52MB total
  unsigned short* xbf  = (unsigned short*)(ws);
  unsigned short* h1   = (unsigned short*)(ws + (((size_t)4)  << 20));
  unsigned short* h2   = (unsigned short*)(ws + (((size_t)12) << 20));
  unsigned short* a0bf = (unsigned short*)(ws + (((size_t)20) << 20));
  unsigned short* a1bf = (unsigned short*)(ws + (((size_t)28) << 20));
  unsigned short* a2bf = (unsigned short*)(ws + (((size_t)44) << 20));

  auto cvt = [&](const float* src, unsigned short* dst, int n) {
    int nq = n / 4;
    cvt_f32_to_bf16<<<(nq + 255) / 256, 256, 0, stream>>>(src, dst, nq);
  };
  cvt(x,  xbf,  4096 * 512);
  cvt(a0, a0bf, 8 * 1024 * 512);
  cvt(a1, a1bf, 8 * 1024 * 1024);
  cvt(a2, a2bf, 8 * 512 * 1024);

  // grid = 32 M-blocks * (OUT/64) N-tiles
  moe_layer<512, 1024, true, false><<<32 * 16, 256, 0, stream>>>(
      xbf, a0bf, wb, b0, h1, nullptr);
  moe_layer<1024, 1024, true, false><<<32 * 16, 256, 0, stream>>>(
      h1, a1bf, wb, b1, h2, nullptr);
  moe_layer<1024, 512, false, true><<<32 * 8, 256, 0, stream>>>(
      h2, a2bf, wb, b2, nullptr, out);
}
